// MinGRULayer_3959959847175
// MI455X (gfx1250) — matrix operand order repaired and checked by reading, not compiled, not run on hardware
//
#include <hip/hip_runtime.h>
#include <hip/hip_bf16.h>
#include <math.h>

// ---------------- problem constants ----------------
#define BB   4
#define TT   4096
#define HH   2048            // hidden dim == K dim
#define MM   (BB*TT)         // 16384 rows
#define G_EPS 0.5f
#define LN_EPS 1e-5f

// chunked scan config
#define NCH  32              // chunks over time
#define CLEN 128             // steps per chunk (NCH*CLEN == TT)
#define NBH  (BB*HH)         // 8192 channels

typedef __bf16 bf16;
typedef __attribute__((ext_vector_type(16))) bf16  v16bf;
typedef __attribute__((ext_vector_type(8)))  bf16  v8bf;
typedef __attribute__((ext_vector_type(8)))  float v8f;

// ---------------- device math helpers ----------------
__device__ __forceinline__ float softplusf(float x) {
    return fmaxf(x, 0.0f) + log1pf(expf(-fabsf(x)));
}
__device__ __forceinline__ float g_logf(float x) {
    return (x >= 0.0f) ? logf(x + G_EPS) : -softplusf(-x);
}

// =====================================================================
// Kernel 1: fused dual GEMM  (k = X@Wz^T, p = X@Wh^T) + log epilogue
//   val[m,n] = logsigmoid(k) + g_log(p),  lc[m,n] = -softplus(k)
// bf16 WMMA 16x16x32, f32 accumulate. Block tile 128(M) x 64(N), BK=32.
// Software-pipelined: next tile's global loads issue before the WMMAs.
// =====================================================================
#define BM 128
#define BN 64
#define BK 32
#define LDSP 48   // padded LDS row stride (bf16 units): 96B, 32B-aligned

__global__ __launch_bounds__(256)
void dual_gemm(const float* __restrict__ X,
               const float* __restrict__ Wz, const float* __restrict__ bz,
               const float* __restrict__ Wh, const float* __restrict__ bh,
               float* __restrict__ val, float* __restrict__ lc)
{
    __shared__ alignas(32) bf16 As[BM][LDSP];
    __shared__ alignas(32) bf16 Bs[2][BN][LDSP];   // [0]=Wz tile, [1]=Wh tile

    const int tid  = threadIdx.x;
    const int lane = tid & 31;
    const int w    = tid >> 5;          // wave 0..7
    const int wm   = w & 3;             // 4 M-groups of 32 rows
    const int wn   = w >> 2;            // 2 N-groups of 32 cols
    const int h2   = lane >> 4;         // half-wave select
    const int i16  = lane & 15;

    // blockIdx.x iterates N-tiles (fast) -> 32 consecutive blocks share one
    // X stripe; weights (32MB) stay L2-resident.
    const int n0 = blockIdx.x * BN;
    const int m0 = blockIdx.y * BM;

    // global->LDS staging mapping
    const int ar   = tid >> 1;          // 0..127  (A row)
    const int ac   = (tid & 1) * 16;    // col segment 0/16
    const int wsel = tid >> 7;          // 0: Wz  1: Wh
    const int wr   = (tid & 127) >> 1;  // 0..63  (W row)
    const int wc   = (tid & 1) * 16;
    const float* __restrict__ Wsrc = wsel ? Wh : Wz;

    const float* __restrict__ gxrow = X    + (size_t)(m0 + ar) * HH + ac;
    const float* __restrict__ gwrow = Wsrc + (size_t)(n0 + wr) * HH + wc;

    v8f accz[2][2], acch[2][2];
#pragma unroll
    for (int a = 0; a < 2; ++a)
#pragma unroll
        for (int b = 0; b < 2; ++b) { accz[a][b] = {}; acch[a][b] = {}; }

    float4 xa[4], wa[4];                 // register staging for pipeline
    auto load_regs = [&](int kbase) {
        const float4* g  = (const float4*)(gxrow + kbase);
        xa[0] = g[0]; xa[1] = g[1]; xa[2] = g[2]; xa[3] = g[3];
        const float4* gw = (const float4*)(gwrow + kbase);
        wa[0] = gw[0]; wa[1] = gw[1]; wa[2] = gw[2]; wa[3] = gw[3];
    };
    auto store_lds = [&]() {
        v8bf lo, hi;
        lo[0]=(bf16)xa[0].x; lo[1]=(bf16)xa[0].y; lo[2]=(bf16)xa[0].z; lo[3]=(bf16)xa[0].w;
        lo[4]=(bf16)xa[1].x; lo[5]=(bf16)xa[1].y; lo[6]=(bf16)xa[1].z; lo[7]=(bf16)xa[1].w;
        hi[0]=(bf16)xa[2].x; hi[1]=(bf16)xa[2].y; hi[2]=(bf16)xa[2].z; hi[3]=(bf16)xa[2].w;
        hi[4]=(bf16)xa[3].x; hi[5]=(bf16)xa[3].y; hi[6]=(bf16)xa[3].z; hi[7]=(bf16)xa[3].w;
        *(v8bf*)(&As[ar][ac])     = lo;
        *(v8bf*)(&As[ar][ac + 8]) = hi;
        v8bf wlo, whi;
        wlo[0]=(bf16)wa[0].x; wlo[1]=(bf16)wa[0].y; wlo[2]=(bf16)wa[0].z; wlo[3]=(bf16)wa[0].w;
        wlo[4]=(bf16)wa[1].x; wlo[5]=(bf16)wa[1].y; wlo[6]=(bf16)wa[1].z; wlo[7]=(bf16)wa[1].w;
        whi[0]=(bf16)wa[2].x; whi[1]=(bf16)wa[2].y; whi[2]=(bf16)wa[2].z; whi[3]=(bf16)wa[2].w;
        whi[4]=(bf16)wa[3].x; whi[5]=(bf16)wa[3].y; whi[6]=(bf16)wa[3].z; whi[7]=(bf16)wa[3].w;
        *(v8bf*)(&Bs[wsel][wr][wc])     = wlo;
        *(v8bf*)(&Bs[wsel][wr][wc + 8]) = whi;
    };

    load_regs(0);
    store_lds();

    for (int kb = 0; kb < HH; kb += BK) {
        __syncthreads();                 // LDS tile ready
        const bool more = (kb + BK) < HH;
        if (more) load_regs(kb + BK);    // overlap global latency with WMMAs

        // ---- build fragments, contiguous k = 16h + s in A and B ----
        v16bf aF[2], bzF[2], bhF[2];
#pragma unroll
        for (int mt = 0; mt < 2; ++mt) {
            const bf16* pa = &As[wm * 32 + mt * 16 + i16][0];
            v8bf lo = *(const v8bf*)(pa + 16 * h2);       // K 16h..16h+7
            v8bf hi = *(const v8bf*)(pa + 16 * h2 + 8);   // K 16h+8..16h+15
            aF[mt] = __builtin_shufflevector(lo, hi,
                      0,1,2,3,4,5,6,7,8,9,10,11,12,13,14,15);
        }
#pragma unroll
        for (int nt = 0; nt < 2; ++nt) {
            const int bn = wn * 32 + nt * 16 + i16;
            bzF[nt] = *(const v16bf*)(&Bs[0][bn][16 * h2]); // K 16h..16h+15
            bhF[nt] = *(const v16bf*)(&Bs[1][bn][16 * h2]);
        }

        // ---- 8 WMMAs per K-step ----
#pragma unroll
        for (int mt = 0; mt < 2; ++mt)
#pragma unroll
            for (int nt = 0; nt < 2; ++nt) {
                accz[mt][nt] = __builtin_amdgcn_wmma_f32_16x16x32_bf16(
                    false, aF[mt], false, bzF[nt], (short)0, accz[mt][nt], false, false);
                acch[mt][nt] = __builtin_amdgcn_wmma_f32_16x16x32_bf16(
                    false, aF[mt], false, bhF[nt], (short)0, acch[mt][nt], false, false);
            }
        __syncthreads();                 // consumers done before overwrite
        if (more) store_lds();
    }

    // ---- fused epilogue: log-space transforms, write val & lc ----
#pragma unroll
    for (int mt = 0; mt < 2; ++mt) {
#pragma unroll
        for (int nt = 0; nt < 2; ++nt) {
            const int n_ = n0 + wn * 32 + nt * 16 + i16;
            const float bzv = bz[n_];
            const float bhv = bh[n_];
#pragma unroll
            for (int r = 0; r < 8; ++r) {
                const int m_ = m0 + wm * 32 + mt * 16 + 8 * h2 + r;
                const float kk = accz[mt][nt][r] + bzv;
                const float pp = acch[mt][nt][r] + bhv;
                const float log_z      = -softplusf(-kk);
                const float log_coeff  = -softplusf(kk);
                const float log_tilde  = g_logf(pp);
                const size_t idx = (size_t)m_ * HH + n_;
                val[idx] = log_z + log_tilde;
                lc[idx]  = log_coeff;
            }
        }
    }
}

// =====================================================================
// Kernel 2: per-row inclusive prefix sum over H (in place: lc -> a_star)
// =====================================================================
__global__ __launch_bounds__(256)
void row_prefix(float* __restrict__ lc)
{
    __shared__ float sc[2][256];
    const int row = blockIdx.x;
    const int t   = threadIdx.x;
    float* p = lc + (size_t)row * HH + t * 8;

    float v[8];
    {
        float4 a = ((const float4*)p)[0];
        float4 b = ((const float4*)p)[1];
        v[0]=a.x; v[1]=a.y; v[2]=a.z; v[3]=a.w;
        v[4]=b.x; v[5]=b.y; v[6]=b.z; v[7]=b.w;
    }
#pragma unroll
    for (int i = 1; i < 8; ++i) v[i] += v[i - 1];

    const float total = v[7];
    sc[0][t] = total;
    __syncthreads();
    int sb = 0;
    for (int off = 1; off < 256; off <<= 1) {
        float x = sc[sb][t];
        if (t >= off) x += sc[sb][t - off];
        sc[sb ^ 1][t] = x;
        sb ^= 1;
        __syncthreads();
    }
    const float excl = sc[sb][t] - total;
#pragma unroll
    for (int i = 0; i < 8; ++i) v[i] += excl;

    ((float4*)p)[0] = make_float4(v[0], v[1], v[2], v[3]);
    ((float4*)p)[1] = make_float4(v[4], v[5], v[6], v[7]);
}

// =====================================================================
// Chunked cumlogsumexp over time (3 passes, 32 chunks x 128 steps)
// =====================================================================

// Pass 1: per-(b,h,chunk) local LSE total -> Lc[c][bh]
__global__ __launch_bounds__(256)
void scan_pass1(const float* __restrict__ val, const float* __restrict__ asx,
                float* __restrict__ Lc)
{
    const int g  = blockIdx.x * 256 + threadIdx.x;   // 0 .. NCH*NBH-1
    const int c  = g >> 13;                          // / NBH
    const int bh = g & (NBH - 1);
    const int b  = bh >> 11;
    const int h  = bh & (HH - 1);
    const size_t base = (size_t)b * TT * HH + h + (size_t)(c * CLEN) * HH;

    float m = -INFINITY, s = 0.0f;
    for (int j = 0; j < CLEN; ++j) {
        const size_t idx = base + (size_t)j * HH;
        const float u  = val[idx] - asx[idx];
        const float nm = fmaxf(m, u);
        s = s * expf(m - nm) + expf(u - nm);
        m = nm;
    }
    Lc[c * NBH + bh] = m + logf(s);
}

// Pass 2: per-(b,h) exclusive LSE prefix over chunks, seeded with g_log(h0)
// (the t'=0 element). In place: Lc[c][bh] becomes the seed for chunk c.
__global__ __launch_bounds__(256)
void scan_pass2(float* __restrict__ Lc, const float* __restrict__ h0)
{
    const int bh = blockIdx.x * 256 + threadIdx.x;   // 0..NBH-1
    const int h  = bh & (HH - 1);
    float run = g_logf(h0[h]);                       // LSE of {u_0}
#pragma unroll 4
    for (int c = 0; c < NCH; ++c) {
        const float t = Lc[c * NBH + bh];
        Lc[c * NBH + bh] = run;                      // exclusive prefix
        const float nm = fmaxf(run, t);
        run = nm + logf(expf(run - nm) + expf(t - nm));
    }
}

// Pass 3: replay each chunk seeded with its prefix; write log_h in place
// over val; accumulate double sum/sumsq partials (chunk 0 adds the t'=0 row).
__global__ __launch_bounds__(256)
void scan_pass3(float* __restrict__ val, const float* __restrict__ asx,
                const float* __restrict__ Lc, double* __restrict__ partials)
{
    __shared__ double ds[256], dq[256];
    const int g  = blockIdx.x * 256 + threadIdx.x;
    const int c  = g >> 13;
    const int bh = g & (NBH - 1);
    const int b  = bh >> 11;
    const int h  = bh & (HH - 1);
    const size_t base = (size_t)b * TT * HH + h + (size_t)(c * CLEN) * HH;

    float m = Lc[c * NBH + bh];                      // seed = LSE(all prior)
    float s = 1.0f;
    double sm = 0.0, sq = 0.0;
    if (c == 0) { sm = (double)m; sq = (double)m * (double)m; } // log_h row 0

    for (int j = 0; j < CLEN; ++j) {
        const size_t idx = base + (size_t)j * HH;
        const float a  = asx[idx];
        const float u  = val[idx] - a;
        const float nm = fmaxf(m, u);
        s = s * expf(m - nm) + expf(u - nm);
        m = nm;
        const float lh = a + m + logf(s);
        val[idx] = lh;                               // log_h[b, t+1, h]
        sm += (double)lh;
        sq += (double)lh * (double)lh;
    }

    ds[threadIdx.x] = sm;
    dq[threadIdx.x] = sq;
    __syncthreads();
    for (int o = 128; o > 0; o >>= 1) {
        if (threadIdx.x < o) {
            ds[threadIdx.x] += ds[threadIdx.x + o];
            dq[threadIdx.x] += dq[threadIdx.x + o];
        }
        __syncthreads();
    }
    if (threadIdx.x == 0) {
        partials[blockIdx.x * 2]     = ds[0];
        partials[blockIdx.x * 2 + 1] = dq[0];
    }
}

// =====================================================================
// Kernel: fixed-order reduction of 1024 partials -> mean, inv_std (ddof=1)
// =====================================================================
__global__ __launch_bounds__(256)
void reduce_stats(const double* __restrict__ partials, float* __restrict__ stats)
{
    __shared__ double ds[256], dq[256];
    const int t = threadIdx.x;
    double S = 0.0, Q = 0.0;
    for (int i = t; i < 1024; i += 256) {            // fixed order per thread
        S += partials[2 * i];
        Q += partials[2 * i + 1];
    }
    ds[t] = S; dq[t] = Q;
    __syncthreads();
    for (int o = 128; o > 0; o >>= 1) {
        if (t < o) { ds[t] += ds[t + o]; dq[t] += dq[t + o]; }
        __syncthreads();
    }
    if (t == 0) {
        const double N    = (double)BB * (double)(TT + 1) * (double)HH;
        const double mean = ds[0] / N;
        const double var  = (dq[0] - ds[0] * ds[0] / N) / (N - 1.0);
        stats[0] = (float)mean;
        stats[1] = (float)(1.0 / sqrt(var));
    }
}

// =====================================================================
// Kernel: h = exp(norm(log_h)); y = h + x; LayerNorm over H (ddof=0)
// =====================================================================
__global__ __launch_bounds__(256)
void finalize(const float* __restrict__ logh, const float* __restrict__ X,
              const float* __restrict__ lnw, const float* __restrict__ lnb,
              const float* __restrict__ stats, float* __restrict__ out)
{
    __shared__ float r1[256], r2[256];
    const int row = blockIdx.x;
    const int t   = threadIdx.x;
    const float mean   = stats[0];
    const float invstd = stats[1];

    const size_t base = (size_t)row * HH + t * 8;
    float y[8];
    float sum = 0.0f, sumsq = 0.0f;
    {
        float4 l0 = ((const float4*)(logh + base))[0];
        float4 l1 = ((const float4*)(logh + base))[1];
        float4 x0 = ((const float4*)(X + base))[0];
        float4 x1 = ((const float4*)(X + base))[1];
        const float lv[8] = {l0.x,l0.y,l0.z,l0.w,l1.x,l1.y,l1.z,l1.w};
        const float xv[8] = {x0.x,x0.y,x0.z,x0.w,x1.x,x1.y,x1.z,x1.w};
#pragma unroll
        for (int i = 0; i < 8; ++i) {
            const float hv = expf((lv[i] - mean) * invstd);
            y[i] = hv + xv[i];
            sum   += y[i];
            sumsq += y[i] * y[i];
        }
    }
    r1[t] = sum; r2[t] = sumsq;
    __syncthreads();
    for (int o = 128; o > 0; o >>= 1) {
        if (t < o) { r1[t] += r1[t + o]; r2[t] += r2[t + o]; }
        __syncthreads();
    }
    const float mu  = r1[0] * (1.0f / HH);
    const float var = r2[0] * (1.0f / HH) - mu * mu;
    const float rs  = rsqrtf(var + LN_EPS);

    float4 w0 = ((const float4*)(lnw + t * 8))[0];
    float4 w1 = ((const float4*)(lnw + t * 8))[1];
    float4 b0 = ((const float4*)(lnb + t * 8))[0];
    float4 b1 = ((const float4*)(lnb + t * 8))[1];
    const float wv[8] = {w0.x,w0.y,w0.z,w0.w,w1.x,w1.y,w1.z,w1.w};
    const float bv[8] = {b0.x,b0.y,b0.z,b0.w,b1.x,b1.y,b1.z,b1.w};

    float o8[8];
#pragma unroll
    for (int i = 0; i < 8; ++i)
        o8[i] = (y[i] - mu) * rs * wv[i] + bv[i];

    ((float4*)(out + base))[0] = make_float4(o8[0], o8[1], o8[2], o8[3]);
    ((float4*)(out + base))[1] = make_float4(o8[4], o8[5], o8[6], o8[7]);
}

// =====================================================================
// host launcher
// =====================================================================
extern "C" void kernel_launch(void* const* d_in, const int* in_sizes, int n_in,
                              void* d_out, int out_size, void* d_ws, size_t ws_size,
                              hipStream_t stream)
{
    const float* X   = (const float*)d_in[0];
    const float* Wz  = (const float*)d_in[1];
    const float* bz  = (const float*)d_in[2];
    const float* Wh  = (const float*)d_in[3];
    const float* bh  = (const float*)d_in[4];
    const float* lnw = (const float*)d_in[5];
    const float* lnb = (const float*)d_in[6];
    const float* h0  = (const float*)d_in[7];
    float* out = (float*)d_out;

    const size_t MH = (size_t)MM * (size_t)HH;
    float*  val      = (float*)d_ws;                 // -> log_h in place
    float*  lcbuf    = val + MH;                     // -> a_star in place
    double* partials = (double*)(val + 2 * MH);      // 1024 pairs
    float*  stats    = (float*)(partials + 2048);    // mean, inv_std
    float*  Lc       = stats + 16;                   // NCH*NBH chunk seeds

    dual_gemm<<<dim3(HH / BN, MM / BM), 256, 0, stream>>>(X, Wz, bz, Wh, bh, val, lcbuf);
    row_prefix<<<MM, 256, 0, stream>>>(lcbuf);
    scan_pass1<<<(NCH * NBH) / 256, 256, 0, stream>>>(val, lcbuf, Lc);
    scan_pass2<<<NBH / 256, 256, 0, stream>>>(Lc, h0);
    scan_pass3<<<(NCH * NBH) / 256, 256, 0, stream>>>(val, lcbuf, Lc, partials);
    reduce_stats<<<1, 256, 0, stream>>>(partials, stats);
    finalize<<<MM, 256, 0, stream>>>(val, X, lnw, lnb, stats, out);
}
